// Encoder_83597243450110
// MI455X (gfx1250) — matrix operand order repaired and checked by reading, not compiled, not run on hardware
//
#include <hip/hip_runtime.h>
#include <hip/hip_bf16.h>
#include <math.h>

// ---------------- problem constants ----------------
#define BB 4
#define TT 12
#define NN 4096
#define FF 32
#define NG 14
#define VW 64                 // fused V width (two F=32 matrices)
#define KSPLIT 4
#define KCHUNK (NN / KSPLIT)  // 1024
#define SCALE_N 4096.0f
#define INV_N   (1.0f / 4096.0f)

typedef _Float16 v16h __attribute__((ext_vector_type(16)));
typedef _Float16 half8 __attribute__((ext_vector_type(8)));
typedef _Float16 half4 __attribute__((ext_vector_type(4)));
typedef float    v8f  __attribute__((ext_vector_type(8)));

// Y partial buffers: [KSPLIT][B][N][VW] f32
#define YSLICE ((size_t)BB * NN * VW)           // 1,048,576 elems per k-split

// d_out layout (fp32): hs [B,T,N,F] then last_h, last_c, last_m [B,N,F]
#define HS_ELEMS ((size_t)BB * TT * NN * FF)    // 6,291,456
#define ST_ELEMS ((size_t)BB * NN * FF)         // 524,288

__device__ __forceinline__ float sigf(float v) { return 1.0f / (1.0f + expf(-v)); }

// ---------------- init kernels ----------------

// adj (fp32, [B,N,N]) -> adjh (f16, scaled by N). 4 elems per thread.
__global__ void __launch_bounds__(256) k_adjconv(const float* __restrict__ adj,
                                                 _Float16* __restrict__ adjh) {
    size_t i = (size_t)blockIdx.x * blockDim.x + threadIdx.x;  // [0, B*N*N/4)
    float4 v = reinterpret_cast<const float4*>(adj)[i];
    half4 h;
    h[0] = (_Float16)(v.x * SCALE_N);
    h[1] = (_Float16)(v.y * SCALE_N);
    h[2] = (_Float16)(v.z * SCALE_N);
    h[3] = (_Float16)(v.w * SCALE_N);
    reinterpret_cast<half4*>(adjh)[i] = h;
}

// init c=m=1, V1t = [x_0 | h0=1] transposed f16: V1t[(b*64+col)*N + n]
__global__ void __launch_bounds__(256) k_init(const float* __restrict__ x,
                                              _Float16* __restrict__ V1t,
                                              float* __restrict__ c,
                                              float* __restrict__ m) {
    int p = blockIdx.x * 256 + threadIdx.x;  // [0, B*VW*N) = 2^20
    int n   = p & (NN - 1);
    int col = (p >> 12) & (VW - 1);
    int b   = p >> 18;
    _Float16 v;
    if (col < FF)
        v = (_Float16)x[(((size_t)b * TT + 0) * NN + n) * FF + col];
    else
        v = (_Float16)1.0f;
    V1t[p] = v;
    if (p < (int)ST_ELEMS) { c[p] = 1.0f; m[p] = 1.0f; }
}

// ---------------- big GEMM: Y[ks] = adjh(16x4096 rows) @ Vt^T  ----------------
// One wave computes a 16(M) x 64(N) tile over a 1024-wide K slice.
// A: adjh row-major [b*N+row][k] (f16, contiguous in k)
// B: Vt [b*64+col][k]          (f16, contiguous in k == transposed V)
// D: Y + ks*YSLICE, row-major [b*N+row][col] f32 partial (plain stores).
__global__ void __launch_bounds__(256) k_gemm(const _Float16* __restrict__ A,
                                              const _Float16* __restrict__ Vt,
                                              float* __restrict__ Y) {
    int wave = blockIdx.x * (blockDim.x >> 5) + (threadIdx.x >> 5);
    int lane = threadIdx.x & 31;
    int ks = wave & (KSPLIT - 1);
    int mt = (wave >> 2) & 255;
    int b  = wave >> 10;

    int mlo = lane & 15;
    int row = mt * 16 + mlo;
    int kbA = (lane < 16) ? 0 : 8;    // A frag: halves j0..7 -> K kbA+j ; j8..15 -> K 16+kbA+j
    int kbB = (lane < 16) ? 0 : 8;    // B frag: halves j0..7 -> K kbB+j ; j8..15 -> K 16+kbB+j

    const _Float16* arow = A + ((size_t)b * NN + row) * NN + (size_t)ks * KCHUNK + kbA;
    const _Float16* bp0 = Vt + ((size_t)(b * VW +  0 + mlo)) * NN + (size_t)ks * KCHUNK + kbB;
    const _Float16* bp1 = Vt + ((size_t)(b * VW + 16 + mlo)) * NN + (size_t)ks * KCHUNK + kbB;
    const _Float16* bp2 = Vt + ((size_t)(b * VW + 32 + mlo)) * NN + (size_t)ks * KCHUNK + kbB;
    const _Float16* bp3 = Vt + ((size_t)(b * VW + 48 + mlo)) * NN + (size_t)ks * KCHUNK + kbB;
    const _Float16* bp[4] = {bp0, bp1, bp2, bp3};

    v8f acc[4] = {};
    for (int kk = 0; kk < KCHUNK; kk += 32) {
        half8 a_lo = *(const half8*)(arow + kk);
        half8 a_hi = *(const half8*)(arow + kk + 16);
        v16h af = __builtin_shufflevector(a_lo, a_hi,
                    0,1,2,3,4,5,6,7,8,9,10,11,12,13,14,15);
#pragma unroll
        for (int ct = 0; ct < 4; ++ct) {
            half8 b_lo = *(const half8*)(bp[ct] + kk);
            half8 b_hi = *(const half8*)(bp[ct] + kk + 16);
            v16h bf = __builtin_shufflevector(b_lo, b_hi,
                        0,1,2,3,4,5,6,7,8,9,10,11,12,13,14,15);
            acc[ct] = __builtin_amdgcn_wmma_f32_16x16x32_f16(
                false, af, false, bf, (short)0, acc[ct], false, false);
        }
    }

    // D layout: VGPR r -> row (lane<16 ? r : 8+r), col = lane&15 (+16*ct)
    float* Yk = Y + (size_t)ks * YSLICE;
    int rbase = mt * 16 + ((lane < 16) ? 0 : 8);
    int cn = lane & 15;
#pragma unroll
    for (int ct = 0; ct < 4; ++ct) {
#pragma unroll
        for (int r = 0; r < 8; ++r) {
            Yk[((size_t)b * NN + rbase + r) * VW + ct * 16 + cn] = acc[ct][r];
        }
    }
}

// ---------------- gate math ----------------
// u = GLU(y @ W[g] + b[g]) at output channel j; y = 32 floats.
__device__ __forceinline__ float wgc_u(const float* y, const float* __restrict__ W,
                                       const float* __restrict__ bias, int g, int j) {
    float l = bias[g * 2 * FF + j];
    float r = bias[g * 2 * FF + FF + j];
    const float* Wg = W + (size_t)g * FF * 2 * FF;
#pragma unroll
    for (int k = 0; k < FF; ++k) {
        float yv = y[k];
        l = fmaf(yv, Wg[k * 2 * FF + j], l);
        r = fmaf(yv, Wg[k * 2 * FF + FF + j], r);
    }
    return l * sigf(r);
}

// Phase A gates: consumes Y1 (= [adj@xt | adj@h] * N, 4 partials), updates c,
// emits V2t = [h_mid | m] (f16 transposed). t==T-1: writes last_c.
__global__ void __launch_bounds__(256) k_gates1(const float* __restrict__ Y1,
                                                const float* __restrict__ W,
                                                const float* __restrict__ bias,
                                                float* __restrict__ c,
                                                const float* __restrict__ m,
                                                _Float16* __restrict__ V2t,
                                                float* __restrict__ out, int t) {
    int tid = blockIdx.x * 256 + threadIdx.x;  // [0, B*N*F)
    int j  = tid & (FF - 1);
    int ng = tid >> 5;                         // b*N + n
    const float* y0 = Y1 + (size_t)ng * VW;
    float y[VW];
#pragma unroll
    for (int k = 0; k < VW; ++k)
        y[k] = (y0[k] + y0[k + YSLICE] + y0[k + 2 * YSLICE] + y0[k + 3 * YSLICE]) * INV_N;

    float u0 = wgc_u(y,      W, bias, 0, j);
    float u1 = wgc_u(y + FF, W, bias, 1, j);
    float u2 = wgc_u(y,      W, bias, 2, j);
    float u3 = wgc_u(y + FF, W, bias, 3, j);
    float u4 = wgc_u(y,      W, bias, 4, j);
    float u5 = wgc_u(y + FF, W, bias, 5, j);
    float u6 = wgc_u(y,      W, bias, 6, j);
    float u7 = wgc_u(y + FF, W, bias, 7, j);

    float fg = sigf(u0 + u1);
    float ig = sigf(u2 + u3);
    float og = sigf(u6 + u7);
    float cn = fg * c[tid] + ig * tanhf(u4 + u5);
    c[tid] = cn;
    float hm = og * tanhf(cn);

    int b = ng >> 12, n = ng & (NN - 1);
    V2t[((size_t)(b * VW) + j) * NN + n]      = (_Float16)hm;
    V2t[((size_t)(b * VW) + FF + j) * NN + n] = (_Float16)m[tid];
    if (t == TT - 1) out[HS_ELEMS + ST_ELEMS + tid] = cn;  // last_c
}

// Phase B gates: consumes Y2 (= [adj@h_mid | adj@m] * N), updates m,
// writes hs[t], V1t = [x_{t+1} | h_new]; t==T-1: last_h, last_m.
__global__ void __launch_bounds__(256) k_gates2(const float* __restrict__ Y2,
                                                const float* __restrict__ W,
                                                const float* __restrict__ bias,
                                                float* __restrict__ m,
                                                const float* __restrict__ x,
                                                _Float16* __restrict__ V1t,
                                                float* __restrict__ out, int t) {
    int tid = blockIdx.x * 256 + threadIdx.x;
    int j  = tid & (FF - 1);
    int ng = tid >> 5;
    const float* y0 = Y2 + (size_t)ng * VW;
    float y[VW];
#pragma unroll
    for (int k = 0; k < VW; ++k)
        y[k] = (y0[k] + y0[k + YSLICE] + y0[k + 2 * YSLICE] + y0[k + 3 * YSLICE]) * INV_N;

    float u8  = wgc_u(y,      W, bias,  8, j);
    float u9  = wgc_u(y + FF, W, bias,  9, j);
    float u10 = wgc_u(y,      W, bias, 10, j);
    float u11 = wgc_u(y + FF, W, bias, 11, j);
    float u12 = wgc_u(y,      W, bias, 12, j);
    float u13 = wgc_u(y + FF, W, bias, 13, j);

    float i2 = sigf(u8 + u9);
    float gg = sigf(u10 + u11);
    float o2 = sigf(u12 + u13);
    float mo = m[tid];
    float mn = i2 * mo + (1.0f - i2) * gg;
    m[tid] = mn;
    float hn = mn * o2;

    int b = ng >> 12, n = ng & (NN - 1);
    out[(((size_t)b * TT + t) * NN + n) * FF + j] = hn;                  // hs[b,t,n,j]
    V1t[((size_t)(b * VW) + FF + j) * NN + n] = (_Float16)hn;            // h for next step
    if (t + 1 < TT)
        V1t[((size_t)(b * VW) + j) * NN + n] =
            (_Float16)x[(((size_t)b * TT + (t + 1)) * NN + n) * FF + j]; // x_{t+1}
    if (t == TT - 1) {
        out[HS_ELEMS + tid]               = hn;  // last_h
        out[HS_ELEMS + 2 * ST_ELEMS + tid] = mn; // last_m
    }
}

// ---------------- launch ----------------
extern "C" void kernel_launch(void* const* d_in, const int* in_sizes, int n_in,
                              void* d_out, int out_size, void* d_ws, size_t ws_size,
                              hipStream_t stream) {
    const float* x    = (const float*)d_in[0];  // [B,T,N,F]
    const float* adj  = (const float*)d_in[1];  // [B,N,N]
    const float* W    = (const float*)d_in[2];  // [NG,F,2F]
    const float* bias = (const float*)d_in[3];  // [NG,2F]
    float* out = (float*)d_out;

    // workspace layout (bytes)
    char* ws = (char*)d_ws;
    _Float16* adjh = (_Float16*)(ws);                              // 134,217,728
    _Float16* V1t  = (_Float16*)(ws + 134217728);                  //   2,097,152
    _Float16* V2t  = (_Float16*)(ws + 136314880);                  //   2,097,152
    float*    Y1   = (float*)   (ws + 138412032);                  //  16,777,216
    float*    Y2   = (float*)   (ws + 155189248);                  //  16,777,216
    float*    c    = (float*)   (ws + 171966464);                  //   2,097,152
    float*    m    = (float*)   (ws + 174063616);                  //   2,097,152

    // adj fp32 -> f16*N (134 MB, fits the 192 MB L2 for all 24 GEMM passes)
    k_adjconv<<<(BB * (size_t)NN * NN / 4) / 256, 256, 0, stream>>>(adj, adjh);
    // c=m=1, V1t=[x0 | 1]
    k_init<<<(BB * VW * NN) / 256, 256, 0, stream>>>(x, V1t, c, m);

    const int gemm_blocks  = (BB * (NN / 16) * KSPLIT) / 8;  // 512 blocks x 8 waves
    const int gate_blocks  = (BB * NN * FF) / 256;           // 2048

    for (int t = 0; t < TT; ++t) {
        k_gemm  <<<gemm_blocks, 256, 0, stream>>>(adjh, V1t, Y1);
        k_gates1<<<gate_blocks, 256, 0, stream>>>(Y1, W, bias, c, m, V2t, out, t);
        k_gemm  <<<gemm_blocks, 256, 0, stream>>>(adjh, V2t, Y2);
        k_gates2<<<gate_blocks, 256, 0, stream>>>(Y2, W, bias, m, x, V1t, out, t);
    }
}
